// DeepThermoMix_70712341561977
// MI455X (gfx1250) — matrix operand order repaired and checked by reading, not compiled, not run on hardware
//
#include <hip/hip_runtime.h>
#include <hip/hip_bf16.h>

typedef _Float16 h8   __attribute__((ext_vector_type(8)));
typedef _Float16 v16h __attribute__((ext_vector_type(16)));
typedef float    v8f  __attribute__((ext_vector_type(8)));
typedef float    f4   __attribute__((ext_vector_type(4)));

#define NROWS   400000
#define MSEG    100000
#define PSTRIDE 132      // pooled row stride in floats (16B aligned rows)
#define LSTRIDE 264      // LDS row stride in halfs (breaks bank conflicts)

__device__ __forceinline__ float softplus_f(float x) {
  // stable log(1+exp(x)) = max(x,0) + log1p(exp(-|x|))
  return fmaxf(x, 0.0f) + log1pf(__expf(-fabsf(x)));
}

__device__ __forceinline__ v16h cat16(h8 lo, h8 hi) {
  v16h r;
#pragma unroll
  for (int i = 0; i < 8; ++i) { r[i] = lo[i]; r[i + 8] = hi[i]; }
  return r;
}

__device__ __forceinline__ h8 cvt8_f32(const float* p) {
  f4 a = *(const f4*)p;
  f4 b = *(const f4*)(p + 4);
  h8 r;
#pragma unroll
  for (int i = 0; i < 4; ++i) { r[i] = (_Float16)a[i]; r[i + 4] = (_Float16)b[i]; }
  return r;
}

// A fragment (16x32 f16). ISA layout: lane L (m=L&15, hi=L>>4) holds
// halfs[0..7] = K = hi*8 .. hi*8+7 and halfs[8..15] = K+16.  rowk = &row[k0].
__device__ __forceinline__ v16h loadA_f32(const float* rowk, int hi) {
  return cat16(cvt8_f32(rowk + hi * 8), cvt8_f32(rowk + hi * 8 + 16));
}
__device__ __forceinline__ v16h loadA_f16(const _Float16* rowk, int hi) {
  return cat16(*(const h8*)(rowk + hi * 8), *(const h8*)(rowk + hi * 8 + 16));
}
// B fragment (32x16 f16). ISA layout: lane (n=L&15, hi=L>>4) holds
// halfs[0..7] = K = hi*8 .. hi*8+7 and halfs[8..15] = K+16.  rowk = &WT[n][k0].
__device__ __forceinline__ v16h loadB_f16(const _Float16* rowk, int hi) {
  const _Float16* p = rowk + hi * 8;
  return cat16(*(const h8*)p, *(const h8*)(p + 16));
}

#define WMMA_F16(A, B, C) \
  __builtin_amdgcn_wmma_f32_16x16x32_f16(false, (A), false, (B), (short)0, (C), false, false)

// ---------------------------------------------------------------- utilities
__global__ void zero_f32(float* p, size_t n) {
  for (size_t i = (size_t)blockIdx.x * blockDim.x + threadIdx.x; i < n;
       i += (size_t)gridDim.x * blockDim.x)
    p[i] = 0.0f;
}

// src: f32 [rows_total, 256];  dst: f16 [256, krows] with dst[n][k]=src[row_off+k][n]
__global__ void wt_transpose(const float* __restrict__ src, _Float16* __restrict__ dst,
                             int krows, int row_off) {
  int idx = blockIdx.x * blockDim.x + threadIdx.x;
  if (idx >= 256 * krows) return;
  int n = idx / krows;
  int k = idx - n * krows;
  dst[(size_t)n * krows + k] = (_Float16)src[(size_t)(row_off + k) * 256 + n];
}

// one block (128 threads) per component row: atomic segment accumulation
__global__ void seg_accum(const float* __restrict__ emb, const float* __restrict__ frac,
                          const int* __restrict__ seg, float* __restrict__ sum,
                          float* __restrict__ cnt) {
  int row = blockIdx.x;
  int t = threadIdx.x;
  int s = seg[row];
  unsafeAtomicAdd(&sum[(size_t)s * PSTRIDE + t], emb[(size_t)row * 128 + t]);
  if (t == 0) {
    unsafeAtomicAdd(&sum[(size_t)s * PSTRIDE + 128], frac[row]);
    unsafeAtomicAdd(&cnt[s], 1.0f);
  }
}

__global__ void pool_finalize(float* __restrict__ pooled, const float* __restrict__ cnt) {
  int m = blockIdx.x;
  int t = threadIdx.x;
  float c = fmaxf(cnt[m], 1.0f);
  if (t < 129) pooled[(size_t)m * PSTRIDE + t] /= c;
}

// ------------------------------------------- fused interaction MLP (M rows)
// pooled[:,0:128] @ iW1[0:128] (WMMA) + pooled[:,128]*iW1[128] (rank-1) + ib1
// -> softplus -> LDS(f16) -> @ iW2 + ib2 -> softplus -> ctx f16 [M,256]
__global__ __launch_bounds__(256) void inter_mlp_kernel(
    const float* __restrict__ pooled, const _Float16* __restrict__ iW1T,
    const float* __restrict__ iW1row128, const float* __restrict__ ib1,
    const _Float16* __restrict__ iW2T, const float* __restrict__ ib2,
    _Float16* __restrict__ ctx) {
  const int tile = blockIdx.x;
  const int lane = threadIdx.x & 31;
  const int wave = threadIdx.x >> 5;
  const int m = lane & 15, hi = lane >> 4;
  const int n0 = wave * 32 + m;
  const int n1 = n0 + 16;

  __shared__ __align__(16) _Float16 lds[16 * LSTRIDE];
  __shared__ float sfr[16];
  if (threadIdx.x < 16)
    sfr[threadIdx.x] = pooled[(size_t)(tile * 16 + threadIdx.x) * PSTRIDE + 128];
  __syncthreads();

  const float* arow = pooled + (size_t)(tile * 16 + m) * PSTRIDE;
  v8f c0 = {}, c1 = {};
#pragma unroll
  for (int kt = 0; kt < 4; ++kt) {
    v16h a = loadA_f32(arow + kt * 32, hi);
    v16h b0 = loadB_f16(iW1T + (size_t)n0 * 128 + kt * 32, hi);
    v16h b1 = loadB_f16(iW1T + (size_t)n1 * 128 + kt * 32, hi);
    c0 = WMMA_F16(a, b0, c0);
    c1 = WMMA_F16(a, b1, c1);
  }
  const float w0 = iW1row128[n0], w1 = iW1row128[n1];
  const float bb0 = ib1[n0], bb1 = ib1[n1];
#pragma unroll
  for (int r = 0; r < 8; ++r) {
    int mr = r + hi * 8;
    float fr = sfr[mr];
    lds[mr * LSTRIDE + n0] = (_Float16)softplus_f(c0[r] + fr * w0 + bb0);
    lds[mr * LSTRIDE + n1] = (_Float16)softplus_f(c1[r] + fr * w1 + bb1);
  }
  __syncthreads();

  v8f d0 = {}, d1 = {};
#pragma unroll
  for (int kt = 0; kt < 8; ++kt) {
    v16h a = loadA_f16(&lds[m * LSTRIDE + kt * 32], hi);
    v16h b0 = loadB_f16(iW2T + (size_t)n0 * 256 + kt * 32, hi);
    v16h b1 = loadB_f16(iW2T + (size_t)n1 * 256 + kt * 32, hi);
    d0 = WMMA_F16(a, b0, d0);
    d1 = WMMA_F16(a, b1, d1);
  }
  const float e0 = ib2[n0], e1 = ib2[n1];
#pragma unroll
  for (int r = 0; r < 8; ++r) {
    int mr = r + hi * 8;
    size_t orow = (size_t)(tile * 16 + mr) * 256;
    ctx[orow + n0] = (_Float16)softplus_f(d0[r] + e0);
    ctx[orow + n1] = (_Float16)softplus_f(d1[r] + e1);
  }
}

// ------------------------------------------------ fused gate MLP (N rows)
// [emb | frac | ctx[seg]] @ gW1 + gb1 -> softplus -> LDS -> @ gW2 + gb2
// -> softplus -> latent f32 (d_out).  frac column = rank-1 with gW1 row 128.
__global__ __launch_bounds__(256) void gate_mlp_kernel(
    const float* __restrict__ emb, const float* __restrict__ frac,
    const int* __restrict__ seg, const _Float16* __restrict__ ctx,
    const _Float16* __restrict__ gW1aT, const _Float16* __restrict__ gW1cT,
    const float* __restrict__ gW1row128, const float* __restrict__ gb1,
    const _Float16* __restrict__ gW2T, const float* __restrict__ gb2,
    float* __restrict__ out) {
  const int tile = blockIdx.x;
  const int lane = threadIdx.x & 31;
  const int wave = threadIdx.x >> 5;
  const int m = lane & 15, hi = lane >> 4;
  const int n0 = wave * 32 + m;
  const int n1 = n0 + 16;

  __shared__ __align__(16) _Float16 lds[16 * LSTRIDE];
  __shared__ int sseg[16];
  __shared__ float sfr[16];
  if (threadIdx.x < 16) {
    int r = tile * 16 + threadIdx.x;
    sseg[threadIdx.x] = seg[r];
    sfr[threadIdx.x] = frac[r];
  }
  __syncthreads();

  const float* arow = emb + (size_t)(tile * 16 + m) * 128;
  v8f c0 = {}, c1 = {};
#pragma unroll
  for (int kt = 0; kt < 4; ++kt) {  // embedding part, K = 0..127
    v16h a = loadA_f32(arow + kt * 32, hi);
    v16h b0 = loadB_f16(gW1aT + (size_t)n0 * 128 + kt * 32, hi);
    v16h b1 = loadB_f16(gW1aT + (size_t)n1 * 128 + kt * 32, hi);
    c0 = WMMA_F16(a, b0, c0);
    c1 = WMMA_F16(a, b1, c1);
  }
  const _Float16* crow = ctx + (size_t)sseg[m] * 256;
#pragma unroll
  for (int kt = 0; kt < 8; ++kt) {  // gathered context part, K = 0..255
    v16h a = loadA_f16(crow + kt * 32, hi);
    v16h b0 = loadB_f16(gW1cT + (size_t)n0 * 256 + kt * 32, hi);
    v16h b1 = loadB_f16(gW1cT + (size_t)n1 * 256 + kt * 32, hi);
    c0 = WMMA_F16(a, b0, c0);
    c1 = WMMA_F16(a, b1, c1);
  }
  const float w0 = gW1row128[n0], w1 = gW1row128[n1];
  const float bb0 = gb1[n0], bb1 = gb1[n1];
#pragma unroll
  for (int r = 0; r < 8; ++r) {
    int mr = r + hi * 8;
    float fr = sfr[mr];
    lds[mr * LSTRIDE + n0] = (_Float16)softplus_f(c0[r] + fr * w0 + bb0);
    lds[mr * LSTRIDE + n1] = (_Float16)softplus_f(c1[r] + fr * w1 + bb1);
  }
  __syncthreads();

  v8f d0 = {}, d1 = {};
#pragma unroll
  for (int kt = 0; kt < 8; ++kt) {
    v16h a = loadA_f16(&lds[m * LSTRIDE + kt * 32], hi);
    v16h b0 = loadB_f16(gW2T + (size_t)n0 * 256 + kt * 32, hi);
    v16h b1 = loadB_f16(gW2T + (size_t)n1 * 256 + kt * 32, hi);
    d0 = WMMA_F16(a, b0, d0);
    d1 = WMMA_F16(a, b1, d1);
  }
  const float e0 = gb2[n0], e1 = gb2[n1];
#pragma unroll
  for (int r = 0; r < 8; ++r) {
    int mr = r + hi * 8;
    size_t orow = (size_t)(tile * 16 + mr) * 256;
    out[orow + n0] = softplus_f(d0[r] + e0);
    out[orow + n1] = softplus_f(d1[r] + e1);
  }
}

// -------------------------------------------------------------------- host
extern "C" void kernel_launch(void* const* d_in, const int* in_sizes, int n_in,
                              void* d_out, int out_size, void* d_ws, size_t ws_size,
                              hipStream_t stream) {
  (void)in_sizes; (void)n_in; (void)out_size; (void)ws_size;
  const float* comp_emb  = (const float*)d_in[0];
  const float* mole_frac = (const float*)d_in[1];
  const int*   seg       = (const int*)d_in[2];
  const float* iW1 = (const float*)d_in[3];
  const float* ib1 = (const float*)d_in[4];
  const float* iW2 = (const float*)d_in[5];
  const float* ib2 = (const float*)d_in[6];
  const float* gW1 = (const float*)d_in[7];
  const float* gb1 = (const float*)d_in[8];
  const float* gW2 = (const float*)d_in[9];
  const float* gb2 = (const float*)d_in[10];

  const int N = NROWS, M = MSEG;

  char* ws = (char*)d_ws;
  size_t off = 0;
  auto alloc = [&](size_t bytes) -> char* {
    char* p = ws + off;
    off = (off + bytes + 255) & ~(size_t)255;
    return p;
  };
  float*    pooled = (float*)alloc((size_t)M * PSTRIDE * sizeof(float));
  float*    cnt    = (float*)alloc((size_t)M * sizeof(float));
  _Float16* iW1T   = (_Float16*)alloc((size_t)256 * 128 * 2);
  _Float16* iW2T   = (_Float16*)alloc((size_t)256 * 256 * 2);
  _Float16* gW1aT  = (_Float16*)alloc((size_t)256 * 128 * 2);
  _Float16* gW1cT  = (_Float16*)alloc((size_t)256 * 256 * 2);
  _Float16* gW2T   = (_Float16*)alloc((size_t)256 * 256 * 2);
  _Float16* ctxh   = (_Float16*)alloc((size_t)M * 256 * 2);

  zero_f32<<<2048, 256, 0, stream>>>(pooled, (size_t)M * PSTRIDE);
  zero_f32<<<512, 256, 0, stream>>>(cnt, (size_t)M);

  wt_transpose<<<(256 * 128 + 255) / 256, 256, 0, stream>>>(iW1, iW1T, 128, 0);
  wt_transpose<<<(256 * 256 + 255) / 256, 256, 0, stream>>>(iW2, iW2T, 256, 0);
  wt_transpose<<<(256 * 128 + 255) / 256, 256, 0, stream>>>(gW1, gW1aT, 128, 0);
  wt_transpose<<<(256 * 256 + 255) / 256, 256, 0, stream>>>(gW1, gW1cT, 256, 129);
  wt_transpose<<<(256 * 256 + 255) / 256, 256, 0, stream>>>(gW2, gW2T, 256, 0);

  seg_accum<<<N, 128, 0, stream>>>(comp_emb, mole_frac, seg, pooled, cnt);
  pool_finalize<<<M, 160, 0, stream>>>(pooled, cnt);

  inter_mlp_kernel<<<M / 16, 256, 0, stream>>>(pooled, iW1T, iW1 + 128 * 256, ib1,
                                               iW2T, ib2, ctxh);
  gate_mlp_kernel<<<N / 16, 256, 0, stream>>>(comp_emb, mole_frac, seg, ctxh, gW1aT,
                                              gW1cT, gW1 + 128 * 256, gb1, gW2T, gb2,
                                              (float*)d_out);

  hipMemcpyAsync((float*)d_out + (size_t)N * 256, mole_frac, (size_t)N * sizeof(float),
                 hipMemcpyDeviceToDevice, stream);
}
